// CSA_84387517432661
// MI455X (gfx1250) — matrix operand order repaired and checked by reading, not compiled, not run on hardware
//
#include <hip/hip_runtime.h>
#include <hip/hip_bf16.h>

// ---------------------------------------------------------------------------
// MI455X (gfx1250, wave32) fused GQA attention block.
// All contractions use v_wmma_f32_16x16x32_f16 (f16 in, f32 accumulate).
// Compute-bound: ~155 GFLOP vs ~150MB working set (fits in 192MB L2).
// ---------------------------------------------------------------------------

typedef __attribute__((ext_vector_type(16))) _Float16 v16h;
typedef __attribute__((ext_vector_type(8)))  _Float16 v8h;
typedef __attribute__((ext_vector_type(8)))  float    v8f;

#define NH_  16
#define NKV_ 4
#define HD_  128
#define DIM_ 2048

// A-operand fragment (16x32 f16), ISA 7.12.2 "16-bit A-Matrix 16x32":
//   lanes 0-15  : row M=lane,    K = {0..7, 16..23}
//   lanes 16-31 : row M=lane-16, K = {8..15, 24..31}
// base points at (row 0, k 0) of the tile; ld = row stride in elements.
__device__ __forceinline__ v16h load_fragA(const _Float16* __restrict__ base,
                                           int ld, int lane) {
  const int r  = lane & 15;
  const int kh = (lane >> 4) << 3;  // 0 or 8
  const _Float16* p = base + (size_t)r * ld + kh;
  const v8h lo = *(const v8h*)(p);        // K = kh .. kh+7
  const v8h hi = *(const v8h*)(p + 16);   // K = kh+16 .. kh+23
  v16h f;
#pragma unroll
  for (int i = 0; i < 8; ++i) { f[i] = lo[i]; f[i + 8] = hi[i]; }
  return f;
}

// B-operand fragment (32x16 f16), per ISA B layout (rows striped across
// lanes; dense analog of the 7.12.4 sparse-B table):
//   lanes 0-15  : column N=lane,    K = {0..7, 16..23}
//   lanes 16-31 : column N=lane-16, K = {8..15, 24..31}
// We load it from B^T storage (N rows, K contiguous): base = (n0, k0).
__device__ __forceinline__ v16h load_fragB(const _Float16* __restrict__ base,
                                           int ld, int lane) {
  const int n  = lane & 15;
  const int kh = (lane >> 4) << 3;  // 0 or 8
  const _Float16* p = base + (size_t)n * ld + kh;
  const v8h lo = *(const v8h*)(p);        // K = kh .. kh+7
  const v8h hi = *(const v8h*)(p + 16);   // K = kh+16 .. kh+23
  v16h f;
#pragma unroll
  for (int i = 0; i < 8; ++i) { f[i] = lo[i]; f[i + 8] = hi[i]; }
  return f;
}

// ---------------------------------------------------------------------------
// fp32 -> fp16 elementwise conversion
// ---------------------------------------------------------------------------
__global__ __launch_bounds__(256)
void k_cvt_f16(const float* __restrict__ s, _Float16* __restrict__ d, size_t n) {
  size_t i = (size_t)blockIdx.x * blockDim.x + threadIdx.x;
  if (i < n) d[i] = (_Float16)s[i];
}

// ---------------------------------------------------------------------------
// C[M,N] = A[M,K] * B[N,K]^T   (A,B f16 row-major K-contiguous, C f32)
// 128 threads = 4 waves arranged 2(M) x 2(N); block tile 64 x 256;
// wave tile 32 x 128 -> per 32-deep k-step: 20 b128 loads feed 16 WMMAs.
// Requires M%64==0, N%256==0, K%32==0.
// ---------------------------------------------------------------------------
__global__ __launch_bounds__(128)
void k_gemm_wmma(const _Float16* __restrict__ A, const _Float16* __restrict__ B,
                 float* __restrict__ C, int M, int N, int K) {
  const int lane = threadIdx.x & 31;
  const int w    = threadIdx.x >> 5;
  const int m0 = blockIdx.y * 64  + (w >> 1) * 32;
  const int n0 = blockIdx.x * 256 + (w & 1)  * 128;
  (void)M;

  v8f acc[2][8];
#pragma unroll
  for (int a = 0; a < 2; ++a)
#pragma unroll
    for (int b = 0; b < 8; ++b)
#pragma unroll
      for (int i = 0; i < 8; ++i) acc[a][b][i] = 0.f;

  const _Float16* Arow0 = A + (size_t)m0 * K;
  const _Float16* Brow0 = B + (size_t)n0 * K;

  for (int k0 = 0; k0 < K; k0 += 32) {
    if (k0 + 32 < K) {                       // gfx1250 global_prefetch_b8
      __builtin_prefetch(Arow0 + k0 + 32, 0, 3);
      __builtin_prefetch(Arow0 + (size_t)16 * K + k0 + 32, 0, 3);
      __builtin_prefetch(Brow0 + k0 + 32, 0, 3);
      __builtin_prefetch(Brow0 + (size_t)64 * K + k0 + 32, 0, 3);
    }
    const v16h a0 = load_fragA(Arow0 + k0, K, lane);
    const v16h a1 = load_fragA(Arow0 + (size_t)16 * K + k0, K, lane);
#pragma unroll
    for (int j = 0; j < 8; ++j) {
      const v16h bf = load_fragB(Brow0 + (size_t)(16 * j) * K + k0, K, lane);
      acc[0][j] = __builtin_amdgcn_wmma_f32_16x16x32_f16(false, a0, false, bf,
                    (short)0, acc[0][j], false, false);
      acc[1][j] = __builtin_amdgcn_wmma_f32_16x16x32_f16(false, a1, false, bf,
                    (short)0, acc[1][j], false, false);
    }
  }

  // C/D layout: vgpr i, lanes 0-15 -> (M=i, N=lane); lanes 16-31 -> (M=8+i).
  const int cn = lane & 15;
  const int rh = (lane >> 4) * 8;
#pragma unroll
  for (int a = 0; a < 2; ++a)
#pragma unroll
    for (int j = 0; j < 8; ++j)
#pragma unroll
      for (int i = 0; i < 8; ++i)
        C[(size_t)(m0 + a * 16 + rh + i) * N + n0 + 16 * j + cn] = acc[a][j][i];
}

// ---------------------------------------------------------------------------
// Fused RMSNorm + RoPE (+ q_gain) + dtype/layout change.
//  q32 [B*T, NH*128]  -> qh [B,NH,T,128]  f16 (normed, roped, gained)
//  k32 [B*T, NKV*128] -> kh [B,NKV,T,128] f16 (normed, roped)
//  v32 [B*T, NKV*128] -> vt [B,NKV,128,T] f16 (transposed for PV B-operand)
// One block per (b,t); 8 waves: waves 0-3 do q heads, waves 4-7 do kv head w-4.
// ---------------------------------------------------------------------------
__global__ __launch_bounds__(256)
void k_normrope(const float* __restrict__ q32, const float* __restrict__ k32,
                const float* __restrict__ v32, const float* __restrict__ qgain,
                _Float16* __restrict__ qh, _Float16* __restrict__ kh,
                _Float16* __restrict__ vt, int T) {
  const int row  = blockIdx.x;          // b*T + t
  const int b    = row / T;
  const int t    = row - b * T;
  const int lane = threadIdx.x & 31;
  const int w    = threadIdx.x >> 5;

  // rope base (T=2048 > 1024 -> rescaled base), inv_freq_j = base^(-2j/128)
  float base = 10000.f;
  if (T > 1024) base = 10000.f * __powf((float)T * (1.f / 1024.f), 128.0f / 126.0f);
  const float nlogb = -__logf(base) * (1.f / 64.f);

  // each lane handles pair indices j0=lane, j1=lane+32 (partners at +64)
  float c0, s0v, c1, s1v;
  {
    const float a0 = __expf((float)lane * nlogb) * (float)t;
    const float a1 = __expf((float)(lane + 32) * nlogb) * (float)t;
    __sincosf(a0, &s0v, &c0);
    __sincosf(a1, &s1v, &c1);
  }
  const float EPS_ = 1.1920928955078125e-07f;

  if (w < 4) {
    const float* qr = q32 + (size_t)row * DIM_;
    for (int h = w; h < NH_; h += 4) {
      const float* p = qr + h * HD_;
      const float x0 = p[lane], x1 = p[lane + 32];
      const float y0 = p[lane + 64], y1 = p[lane + 96];
      float ss = x0 * x0 + x1 * x1 + y0 * y0 + y1 * y1;
#pragma unroll
      for (int m = 1; m < 32; m <<= 1) ss += __shfl_xor(ss, m);
      const float g = rsqrtf(ss * (1.f / 128.f) + EPS_) * qgain[h];
      _Float16* o = qh + ((size_t)(b * NH_ + h) * T + t) * HD_;
      o[lane]      = (_Float16)((x0 * c0 + y0 * s0v) * g);
      o[lane + 64] = (_Float16)((y0 * c0 - x0 * s0v) * g);
      o[lane + 32] = (_Float16)((x1 * c1 + y1 * s1v) * g);
      o[lane + 96] = (_Float16)((y1 * c1 - x1 * s1v) * g);
    }
  } else {
    const int h = w - 4;  // kv head
    {
      const float* p = k32 + (size_t)row * (NKV_ * HD_) + h * HD_;
      const float x0 = p[lane], x1 = p[lane + 32];
      const float y0 = p[lane + 64], y1 = p[lane + 96];
      float ss = x0 * x0 + x1 * x1 + y0 * y0 + y1 * y1;
#pragma unroll
      for (int m = 1; m < 32; m <<= 1) ss += __shfl_xor(ss, m);
      const float g = rsqrtf(ss * (1.f / 128.f) + EPS_);
      _Float16* o = kh + ((size_t)(b * NKV_ + h) * T + t) * HD_;
      o[lane]      = (_Float16)((x0 * c0 + y0 * s0v) * g);
      o[lane + 64] = (_Float16)((y0 * c0 - x0 * s0v) * g);
      o[lane + 32] = (_Float16)((x1 * c1 + y1 * s1v) * g);
      o[lane + 96] = (_Float16)((y1 * c1 - x1 * s1v) * g);
    }
    {
      const float* pv = v32 + (size_t)row * (NKV_ * HD_) + h * HD_;
      _Float16* vo = vt + (size_t)(b * NKV_ + h) * HD_ * T + t;
#pragma unroll
      for (int q = 0; q < 4; ++q)
        vo[(size_t)(lane + 32 * q) * T] = (_Float16)pv[lane + 32 * q];
    }
  }
}

// ---------------------------------------------------------------------------
// Causal flash attention. Grid (T/64, NH, B), 128 threads = 4 waves.
// Each wave owns 16 query rows: S = Q(16x128) K^T, online softmax per
// C-layout row slot, P bounced through per-wave LDS (C->A layout), then
// acc(16x128) += P(16x32) V(32x128). GQA: kv head = h >> 2.
// ---------------------------------------------------------------------------
__global__ __launch_bounds__(128)
void k_attn(const _Float16* __restrict__ qh, const _Float16* __restrict__ kh,
            const _Float16* __restrict__ vt, _Float16* __restrict__ oh, int T) {
  __shared__ _Float16 sP[4][16 * 32];   // 4KB, one 16x32 P tile per wave
  const int lane = threadIdx.x & 31;
  const int w    = threadIdx.x >> 5;
  const int h    = blockIdx.y;
  const int b    = blockIdx.z;
  const int kvh  = h >> 2;
  const int q0   = blockIdx.x * 64 + w * 16;

  const _Float16* Qb = qh + (size_t)(b * NH_ + h) * T * HD_;
  const _Float16* Kb = kh + (size_t)(b * NKV_ + kvh) * T * HD_;
  const _Float16* Vb = vt + (size_t)(b * NKV_ + kvh) * HD_ * T;
  _Float16* myP = sP[w];

  v16h qf[4];
#pragma unroll
  for (int kc = 0; kc < 4; ++kc)
    qf[kc] = load_fragA(Qb + (size_t)q0 * HD_ + kc * 32, HD_, lane);

  v8f acc[8];
#pragma unroll
  for (int j = 0; j < 8; ++j)
#pragma unroll
    for (int i = 0; i < 8; ++i) acc[j][i] = 0.f;
  float mi[8], li[8];
#pragma unroll
  for (int i = 0; i < 8; ++i) { mi[i] = -1e30f; li[i] = 0.f; }

  const float sc  = 0.08838834764831845f;  // 1/sqrt(128)
  const int ncol  = lane & 15;
  const int rh    = (lane >> 4) * 8;

  const int nmax = (q0 + 15) >> 5;         // causal: 32-key blocks
  for (int n32 = 0; n32 <= nmax; ++n32) {
    const int kb0 = n32 * 32;
    v8f S0, S1;
#pragma unroll
    for (int i = 0; i < 8; ++i) { S0[i] = 0.f; S1[i] = 0.f; }
#pragma unroll
    for (int kc = 0; kc < 4; ++kc) {
      const v16h kf0 = load_fragB(Kb + (size_t)kb0 * HD_ + kc * 32, HD_, lane);
      S0 = __builtin_amdgcn_wmma_f32_16x16x32_f16(false, qf[kc], false, kf0,
             (short)0, S0, false, false);
      const v16h kf1 = load_fragB(Kb + (size_t)(kb0 + 16) * HD_ + kc * 32, HD_, lane);
      S1 = __builtin_amdgcn_wmma_f32_16x16x32_f16(false, qf[kc], false, kf1,
             (short)0, S1, false, false);
    }

    // per C-layout row slot i: row = q0 + rh + i; 16-lane butterfly reductions
#pragma unroll
    for (int i = 0; i < 8; ++i) {
      const int qrow = q0 + rh + i;
      float s0 = (kb0 + ncol      <= qrow) ? S0[i] * sc : -1e30f;
      float s1 = (kb0 + 16 + ncol <= qrow) ? S1[i] * sc : -1e30f;
      float tm = fmaxf(s0, s1);
#pragma unroll
      for (int m = 1; m < 16; m <<= 1) tm = fmaxf(tm, __shfl_xor(tm, m));
      const float nm = fmaxf(mi[i], tm);
      const float p0 = __expf(s0 - nm);
      const float p1 = __expf(s1 - nm);
      float rs = p0 + p1;
#pragma unroll
      for (int m = 1; m < 16; m <<= 1) rs += __shfl_xor(rs, m);
      const float corr = __expf(mi[i] - nm);
      li[i] = li[i] * corr + rs;
      mi[i] = nm;
#pragma unroll
      for (int j = 0; j < 8; ++j) acc[j][i] *= corr;
      // stage P in C layout
      myP[(rh + i) * 32 + ncol]      = (_Float16)p0;
      myP[(rh + i) * 32 + 16 + ncol] = (_Float16)p1;
    }

    // per-wave LDS ordering (no block barrier: waves diverge in trip count)
    asm volatile("s_wait_dscnt 0" ::: "memory");

    // reload P in A layout (16 rows x 32 kv)
    const _Float16* pp = myP + (lane & 15) * 32 + ((lane >> 4) << 3);
    const v8h plo = *(const v8h*)(pp);
    const v8h phi = *(const v8h*)(pp + 16);
    v16h pa;
#pragma unroll
    for (int i = 0; i < 8; ++i) { pa[i] = plo[i]; pa[i + 8] = phi[i]; }

#pragma unroll
    for (int j = 0; j < 8; ++j) {
      const v16h vf = load_fragB(Vb + (size_t)(j * 16) * T + kb0, T, lane);
      acc[j] = __builtin_amdgcn_wmma_f32_16x16x32_f16(false, pa, false, vf,
                 (short)0, acc[j], false, false);
    }
  }

  float inv[8];
#pragma unroll
  for (int i = 0; i < 8; ++i) inv[i] = 1.f / li[i];
#pragma unroll
  for (int j = 0; j < 8; ++j)
#pragma unroll
    for (int i = 0; i < 8; ++i)
      oh[(size_t)(b * T + q0 + rh + i) * DIM_ + h * HD_ + j * 16 + ncol] =
          (_Float16)(acc[j][i] * inv[i]);
}

// ---------------------------------------------------------------------------
// Host-side orchestration. Workspace layout (liveness-overlapped, peak 84MB):
//   [0,8)MB    wp_h (live whole pipeline)
//   [8,36)MB   region A: x_h/wq_h/wk_h/wv_h  -> later q_h/k_h/v_t
//   [36,84)MB  region B: q32/k32/v32         -> later o_h
// ---------------------------------------------------------------------------
extern "C" void kernel_launch(void* const* d_in, const int* in_sizes, int n_in,
                              void* d_out, int out_size, void* d_ws, size_t ws_size,
                              hipStream_t stream) {
  (void)in_sizes; (void)n_in; (void)out_size; (void)ws_size;
  const float* x  = (const float*)d_in[0];
  const float* Wq = (const float*)d_in[1];
  const float* Wk = (const float*)d_in[2];
  const float* Wv = (const float*)d_in[3];
  const float* Wp = (const float*)d_in[4];
  const float* qg = (const float*)d_in[5];

  const int B = 2, T = 2048;
  const int M = B * T;                 // 4096
  const int KD = NKV_ * HD_;           // 512
  const size_t MB = (size_t)1 << 20;

  char* ws = (char*)d_ws;
  _Float16* wp_h = (_Float16*)(ws);                   // 8 MB
  char* rA = ws + 8 * MB;
  _Float16* x_h  = (_Float16*)(rA);                   // 16 MB
  _Float16* wq_h = (_Float16*)(rA + 16 * MB);         // 8 MB
  _Float16* wk_h = (_Float16*)(rA + 24 * MB);         // 2 MB
  _Float16* wv_h = (_Float16*)(rA + 26 * MB);         // 2 MB
  char* rB = ws + 36 * MB;
  float* q32 = (float*)(rB);                          // 32 MB
  float* k32 = (float*)(rB + 32 * MB);                // 8 MB
  float* v32 = (float*)(rB + 40 * MB);                // 8 MB (ends at 84MB)
  // stage-3 outputs overlap dead region A
  _Float16* q_h = (_Float16*)(rA);                    // 16 MB
  _Float16* k_h = (_Float16*)(rA + 16 * MB);          // 4 MB
  _Float16* v_t = (_Float16*)(rA + 20 * MB);          // 4 MB
  // stage-4 output overlaps dead q32
  _Float16* o_h = (_Float16*)(rB);                    // 16 MB

  // 1) fp32 -> fp16 staging
  auto g1 = [](size_t n) { return dim3((unsigned)((n + 255) / 256)); };
  k_cvt_f16<<<g1((size_t)M * DIM_),    256, 0, stream>>>(x,  x_h,  (size_t)M * DIM_);
  k_cvt_f16<<<g1((size_t)DIM_ * DIM_), 256, 0, stream>>>(Wq, wq_h, (size_t)DIM_ * DIM_);
  k_cvt_f16<<<g1((size_t)KD * DIM_),   256, 0, stream>>>(Wk, wk_h, (size_t)KD * DIM_);
  k_cvt_f16<<<g1((size_t)KD * DIM_),   256, 0, stream>>>(Wv, wv_h, (size_t)KD * DIM_);
  k_cvt_f16<<<g1((size_t)DIM_ * DIM_), 256, 0, stream>>>(Wp, wp_h, (size_t)DIM_ * DIM_);

  // 2) QKV projections (WMMA GEMMs); block tile 64(M) x 256(N)
  k_gemm_wmma<<<dim3(DIM_ / 256, M / 64), 128, 0, stream>>>(x_h, wq_h, q32, M, DIM_, DIM_);
  k_gemm_wmma<<<dim3(KD   / 256, M / 64), 128, 0, stream>>>(x_h, wk_h, k32, M, KD,   DIM_);
  k_gemm_wmma<<<dim3(KD   / 256, M / 64), 128, 0, stream>>>(x_h, wv_h, v32, M, KD,   DIM_);

  // 3) RMSNorm + RoPE + gain + layout change
  k_normrope<<<dim3((unsigned)M), 256, 0, stream>>>(q32, k32, v32, qg, q_h, k_h, v_t, T);

  // 4) causal flash attention (WMMA)
  k_attn<<<dim3(T / 64, NH_, B), 128, 0, stream>>>(q_h, k_h, v_t, o_h, T);

  // 5) output projection (WMMA GEMM) -> f32 d_out
  k_gemm_wmma<<<dim3(DIM_ / 256, M / 64), 128, 0, stream>>>(o_h, wp_h, (float*)d_out, M, DIM_, DIM_);
}
